// Mini_Batch_ERGCN_7627861918260
// MI455X (gfx1250) — matrix operand order repaired and checked by reading, compiled, not run on hardware
//
#include <hip/hip_runtime.h>

// ---------------- problem sizes (static, from reference) ----------------
#define EDIM   128
#define CDIM   32
#define RREL   64
#define BDIM   16
#define NN1    8192
#define NN2    8192
#define MM1    4096
#define MM2    4096
#define NHH    8192
#define MOUT   2048
#define NNZ1   131072
#define NNZ2   131072
#define NNZ3   65536

#define RCHUNK 16
#define NCHUNK (RREL / RCHUNK)

#define APAD   8                 // bf16 elements of padding per LDS A row
#define AROW   (EDIM + APAD)     // 136 elements = 272 B row stride (bank-conflict free)

typedef __attribute__((ext_vector_type(16))) __bf16          v16bf;
typedef __attribute__((ext_vector_type(16))) unsigned short  v16us;
typedef __attribute__((ext_vector_type(8)))  unsigned short  v8us;
typedef __attribute__((ext_vector_type(8)))  float           v8f;

union ABfrag { v8us h[2]; v16us u; v16bf b; };

static __device__ __forceinline__ unsigned short f32_to_bf16(float f) {
    unsigned int u = __float_as_uint(f);
    unsigned int r = (u + 0x7FFFu + ((u >> 16) & 1u)) >> 16;   // RNE
    return (unsigned short)r;
}

// ---------------- utility: zero fill ----------------
__global__ void k_zero_f32(float* __restrict__ p, size_t n) {
    size_t i = (size_t)blockIdx.x * blockDim.x + threadIdx.x;
    size_t stride = (size_t)gridDim.x * blockDim.x;
    for (; i < n; i += stride) p[i] = 0.0f;
}

// ---------------- pack fp32 -> bf16 (8 elements / thread-iter) ----------------
__global__ void k_pack_bf16(const float* __restrict__ src,
                            unsigned short* __restrict__ dst, size_t n8) {
    size_t i = (size_t)blockIdx.x * blockDim.x + threadIdx.x;
    size_t stride = (size_t)gridDim.x * blockDim.x;
    for (; i < n8; i += stride) {
        const float4 a = ((const float4*)src)[i * 2];
        const float4 b = ((const float4*)src)[i * 2 + 1];
        v8us o;
        o[0] = f32_to_bf16(a.x); o[1] = f32_to_bf16(a.y);
        o[2] = f32_to_bf16(a.z); o[3] = f32_to_bf16(a.w);
        o[4] = f32_to_bf16(b.x); o[5] = f32_to_bf16(b.y);
        o[6] = f32_to_bf16(b.z); o[7] = f32_to_bf16(b.w);
        ((v8us*)dst)[i] = o;
    }
}

// ---------------- compose W1^T: w1t[r][h][e] = sum_b comp1[r,b]*bases1[b,e,h] (bf16) ----
__global__ void k_compose_w1(const float* __restrict__ comp1,
                             const float* __restrict__ bases1,
                             unsigned short* __restrict__ w1t) {
    int idx = blockIdx.x * blockDim.x + threadIdx.x;
    if (idx >= RREL * EDIM * EDIM) return;
    int r   = idx / (EDIM * EDIM);
    int rem = idx % (EDIM * EDIM);
    int h   = rem / EDIM;
    int e   = rem % EDIM;
    float acc = 0.0f;
#pragma unroll
    for (int b = 0; b < BDIM; ++b)
        acc += comp1[r * BDIM + b] * bases1[((size_t)b * EDIM + e) * EDIM + h];
    w1t[idx] = f32_to_bf16(acc);
}

// ---------------- compose W2^T: w2t[r][c][h] = sum_b comp2[r,b]*bases2[b,h,c] (bf16) ----
__global__ void k_compose_w2(const float* __restrict__ comp2,
                             const float* __restrict__ bases2,
                             unsigned short* __restrict__ w2t) {
    int idx = blockIdx.x * blockDim.x + threadIdx.x;
    if (idx >= RREL * CDIM * EDIM) return;
    int r   = idx / (CDIM * EDIM);
    int rem = idx % (CDIM * EDIM);
    int c   = rem / EDIM;
    int h   = rem % EDIM;
    float acc = 0.0f;
#pragma unroll
    for (int b = 0; b < BDIM; ++b)
        acc += comp2[r * BDIM + b] * bases2[((size_t)b * EDIM + h) * CDIM + c];
    w2t[idx] = f32_to_bf16(acc);
}

// ---------------- COO scatter (layer 1): T[(r-rbase), row, e] += val * X[n, e] ---------
__global__ void k_scatter_l1(const float* __restrict__ val,
                             const int*   __restrict__ row,
                             const int*   __restrict__ col,
                             const float* __restrict__ X,
                             float* __restrict__ T,
                             int nSrc, int mRows, int rbase) {
    int i = blockIdx.x;
    int c = col[i];
    int r = c / nSrc;
    if (r < rbase || r >= rbase + RCHUNK) return;
    int n = c - r * nSrc;
    int m = row[i];
    float v = val[i];
    int e = threadIdx.x;
    atomicAdd(&T[(((size_t)(r - rbase) * mRows + m) * EDIM) + e],
              v * X[(size_t)n * EDIM + e]);
}

// ---------------- COO scatter (layer 2): gathers h1[H_idx[n]] on the fly -------------
__global__ void k_scatter_l3(const float* __restrict__ val,
                             const int*   __restrict__ row,
                             const int*   __restrict__ col,
                             const float* __restrict__ h1,
                             const int*   __restrict__ Hidx,
                             float* __restrict__ T,
                             int rbase) {
    int i = blockIdx.x;
    int c = col[i];
    int r = c / NHH;
    if (r < rbase || r >= rbase + RCHUNK) return;
    int n = c - r * NHH;
    int src = Hidx[n];
    int m = row[i];
    float v = val[i];
    int e = threadIdx.x;
    atomicAdd(&T[(((size_t)(r - rbase) * MOUT + m) * EDIM) + e],
              v * h1[(size_t)src * EDIM + e]);
}

// ---------------- batched-accumulate WMMA GEMM -------------------------------------
// Cout[m, n] += sum_{rl<RCHUNK} sum_{k<128} Tb[rl, m, k] * Wt[rl, n, k]
// Tb  : bf16 [RCHUNK][Mrows][128]     (A matrices, row-major)
// Wt  : bf16 [RCHUNK][Ncols][128]     (B matrices, pre-transposed: [n][k])
// Cout: fp32 [Mrows][Ncols]
//
// 256 threads = 8 waves. Each wave owns a 16x32 C tile (two accumulators) so
// every A fragment feeds two WMMAs. Waves are grouped by M-tile and share one
// 16x128 A tile per relation step via LDS (double-buffered, one barrier per
// step). All 12 fragments of a step are preloaded, then 8 WMMAs issue
// back-to-back (independent accumulators -> no RAW hazards).
__global__ __launch_bounds__(256)
void k_gemm_acc(const unsigned short* __restrict__ Tb,
                const unsigned short* __restrict__ Wt,
                float* __restrict__ Cout,
                int Mrows, int Ncols) {
    const int nPairs  = Ncols >> 5;                     // waves/group: 4 (L1) or 1 (L2)
    const int gPerBlk = 8 / nPairs;                     // 2 or 8

    __shared__ unsigned short As[2][8][16 * AROW];      // 2 buf x up to 8 groups x 4.25KB

    const int wave  = threadIdx.x >> 5;
    const int lane  = threadIdx.x & 31;
    const int group = wave / nPairs;
    const int nP    = wave - group * nPairs;
    const int mTile = blockIdx.x * gPerBlk + group;
    const int mBase = mTile << 4;
    const int nBase = nP << 5;                          // 32 columns per wave
    const int lmod  = lane & 15;
    const int lhi   = lane >> 4;                        // 0: lanes 0-15, 1: lanes 16-31

    const int threadsPerGroup = nPairs << 5;
    const int tidInGroup      = nP * 32 + lane;

    // ---- load existing C tiles (accumulated across relation chunks) ----
    // D/C layout: VGPR v, lane l -> M = v + (l>=16 ? 8 : 0), N = l%16
    v8f acc0, acc1;
    const int crow = mBase + lhi * 8;
#pragma unroll
    for (int v = 0; v < 8; ++v) {
        acc0[v] = Cout[(size_t)(crow + v) * Ncols + nBase + lmod];
        acc1[v] = Cout[(size_t)(crow + v) * Ncols + nBase + 16 + lmod];
    }

    // cooperative A-tile loader: 16 rows x 16 chunks of 8 bf16 (16B each)
    auto load_tile = [&](int buf, int rl) {
        const unsigned short* src = Tb + ((size_t)rl * Mrows + mBase) * EDIM;
        unsigned short* dst = As[buf][group];
        for (int c = tidInGroup; c < 256; c += threadsPerGroup) {
            const int rrow = c >> 4, chunk = c & 15;
            const uint4 v = *(const uint4*)(src + (size_t)rrow * EDIM + chunk * 8);
            *(uint4*)(dst + rrow * AROW + chunk * 8) = v;
        }
    };

    load_tile(0, 0);

    for (int rl = 0; rl < RCHUNK; ++rl) {
        __syncthreads();                                 // tile rl visible; prev buf free
        if (rl + 1 < RCHUNK) load_tile((rl + 1) & 1, rl + 1);

        const unsigned short* arow = &As[rl & 1][group][lmod * AROW];
        const unsigned short* Bw0  = Wt + ((size_t)rl * Ncols + nBase + lmod) * EDIM;
        const unsigned short* Bw1  = Bw0 + (size_t)16 * EDIM;

        // ---- preload all fragments of this relation step ----
        ABfrag a[4], b0[4], b1[4];
#pragma unroll
        for (int ks = 0; ks < 4; ++ks) {                 // K = 128 in steps of 32
            // A fragment (16x32 bf16): lane row M=l%16,
            //   lanes 0-15: K = ks*32 + 0..15; lanes 16-31: K = ks*32 + 16..31
            const int kb = ks * 32 + lhi * 16;
            a[ks].h[0] = *(const v8us*)(arow + kb);
            a[ks].h[1] = *(const v8us*)(arow + kb + 8);
            // B fragments (32x16 bf16): lane col N=l%16,
            //   lanes 0-15: K = ks*32 + 0..15; lanes 16-31: K = ks*32 + 16..31
            const int kbb = ks * 32 + lhi * 16;
            b0[ks].h[0] = *(const v8us*)(Bw0 + kbb);
            b0[ks].h[1] = *(const v8us*)(Bw0 + kbb + 8);
            b1[ks].h[0] = *(const v8us*)(Bw1 + kbb);
            b1[ks].h[1] = *(const v8us*)(Bw1 + kbb + 8);
        }
        // ---- 8 back-to-back WMMAs ----
#pragma unroll
        for (int ks = 0; ks < 4; ++ks) {
            acc0 = __builtin_amdgcn_wmma_f32_16x16x32_bf16(
                false, a[ks].b, false, b0[ks].b, (short)0, acc0, false, false);
            acc1 = __builtin_amdgcn_wmma_f32_16x16x32_bf16(
                false, a[ks].b, false, b1[ks].b, (short)0, acc1, false, false);
        }
    }

#pragma unroll
    for (int v = 0; v < 8; ++v) {
        Cout[(size_t)(crow + v) * Ncols + nBase + lmod]      = acc0[v];
        Cout[(size_t)(crow + v) * Ncols + nBase + 16 + lmod] = acc1[v];
    }
}

// ---------------- bias + relu over h1 ----------------
__global__ void k_bias_relu(float* __restrict__ h1, const float* __restrict__ bias, size_t n) {
    size_t i = (size_t)blockIdx.x * blockDim.x + threadIdx.x;
    size_t stride = (size_t)gridDim.x * blockDim.x;
    for (; i < n; i += stride) {
        float v = h1[i] + bias[i % EDIM];
        h1[i] = v > 0.0f ? v : 0.0f;
    }
}

__global__ void k_bias2(float* __restrict__ out, const float* __restrict__ bias, size_t n) {
    size_t i = (size_t)blockIdx.x * blockDim.x + threadIdx.x;
    size_t stride = (size_t)gridDim.x * blockDim.x;
    for (; i < n; i += stride) out[i] += bias[i % CDIM];
}

// =====================================================================================
extern "C" void kernel_launch(void* const* d_in, const int* in_sizes, int n_in,
                              void* d_out, int out_size, void* d_ws, size_t ws_size,
                              hipStream_t stream) {
    (void)in_sizes; (void)n_in; (void)out_size; (void)ws_size;

    const float* X_nb1  = (const float*)d_in[0];
    const float* X_nb2  = (const float*)d_in[1];
    const float* comp1  = (const float*)d_in[2];
    const float* bases1 = (const float*)d_in[3];
    const float* comp2  = (const float*)d_in[4];
    const float* bases2 = (const float*)d_in[5];
    const float* bias1  = (const float*)d_in[6];
    const float* bias2  = (const float*)d_in[7];
    const float* A1v    = (const float*)d_in[8];
    const float* A2v    = (const float*)d_in[9];
    const float* A3v    = (const float*)d_in[10];
    const int*   A1r    = (const int*)d_in[11];
    const int*   A1c    = (const int*)d_in[12];
    const int*   A2r    = (const int*)d_in[13];
    const int*   A2c    = (const int*)d_in[14];
    const int*   A3r    = (const int*)d_in[15];
    const int*   A3c    = (const int*)d_in[16];
    const int*   Hidx   = (const int*)d_in[17];
    float* out = (float*)d_out;

    // ---- workspace layout ----
    char* ws = (char*)d_ws;
    const size_t T_ELTS  = (size_t)RCHUNK * MM1 * EDIM;             // 8M floats (32 MB)
    float*          T   = (float*)ws;
    unsigned short* Tb  = (unsigned short*)(ws + T_ELTS * 4);       // 16 MB bf16 mirror
    float*          h1  = (float*)((char*)Tb + T_ELTS * 2);         // 4 MB
    unsigned short* w1t = (unsigned short*)((char*)h1 + (size_t)(MM1 + MM2) * EDIM * 4);
    unsigned short* w2t = w1t + (size_t)RREL * EDIM * EDIM;

    const size_t H1_ELTS = (size_t)(MM1 + MM2) * EDIM;

    // 1) compose basis-decomposed weights (bf16, transposed for WMMA-B loads)
    k_compose_w1<<<(RREL * EDIM * EDIM + 255) / 256, 256, 0, stream>>>(comp1, bases1, w1t);
    k_compose_w2<<<(RREL * CDIM * EDIM + 255) / 256, 256, 0, stream>>>(comp2, bases2, w2t);

    // 2) init accumulators
    k_zero_f32<<<2048, 256, 0, stream>>>(h1, H1_ELTS);
    k_zero_f32<<<256, 256, 0, stream>>>(out, (size_t)MOUT * CDIM);

    // 3) layer 1: per depth & relation chunk: scatter -> pack bf16 -> WMMA-GEMM accumulate
    for (int phase = 0; phase < 2; ++phase) {
        const float* X   = phase ? X_nb2 : X_nb1;
        const float* val = phase ? A2v : A1v;
        const int*   rr  = phase ? A2r : A1r;
        const int*   cc  = phase ? A2c : A1c;
        const int    nsrc = phase ? NN2 : NN1;
        float* hout = h1 + (size_t)phase * MM1 * EDIM;
        for (int ch = 0; ch < NCHUNK; ++ch) {
            k_zero_f32<<<8192, 256, 0, stream>>>(T, T_ELTS);
            k_scatter_l1<<<NNZ1, EDIM, 0, stream>>>(val, rr, cc, X, T, nsrc, MM1, ch * RCHUNK);
            k_pack_bf16<<<2048, 256, 0, stream>>>(T, Tb, T_ELTS / 8);
            const int blocks1 = (MM1 / 16) / 2;                     // 2 groups (4 waves ea)/blk
            k_gemm_acc<<<blocks1, 256, 0, stream>>>(
                Tb, w1t + (size_t)ch * RCHUNK * EDIM * EDIM, hout, MM1, EDIM);
        }
    }

    // 4) bias + relu on hidden states
    k_bias_relu<<<2048, 256, 0, stream>>>(h1, bias1, H1_ELTS);

    // 5) layer 2: scatter (H_idx gather fused) -> pack -> WMMA-GEMM accumulate into out
    const size_t T3_ELTS = (size_t)RCHUNK * MOUT * EDIM;            // 4M floats
    for (int ch = 0; ch < NCHUNK; ++ch) {
        k_zero_f32<<<4096, 256, 0, stream>>>(T, T3_ELTS);
        k_scatter_l3<<<NNZ3, EDIM, 0, stream>>>(A3v, A3r, A3c, h1, Hidx, T, ch * RCHUNK);
        k_pack_bf16<<<1024, 256, 0, stream>>>(T, Tb, T3_ELTS / 8);
        const int blocks2 = (MOUT / 16) / 8;                        // 8 groups (1 wave ea)/blk
        k_gemm_acc<<<blocks2, 256, 0, stream>>>(
            Tb, w2t + (size_t)ch * RCHUNK * CDIM * EDIM, out, MOUT, CDIM);
    }

    // 6) output bias
    k_bias2<<<256, 256, 0, stream>>>(out, bias2, (size_t)MOUT * CDIM);
}
